// MyModel_87522843559735
// MI455X (gfx1250) — matrix operand order repaired and checked by reading, not compiled, not run on hardware
//
#include <hip/hip_runtime.h>

// Problem constants (from reference): N=200 LSTMs, B=100, T=256, D=5, H=20
#define NLSTM 200
#define BATCH 100
#define SEQT  256
#define DIN   5
#define HID   20
#define G4H   80      // 4*H gate width
#define NH    4000    // N*H concat width
#define ROWS  25600   // B*T rows of the dense GEMM
#define XELTS (BATCH * SEQT * DIN)   // 128000

typedef __attribute__((ext_vector_type(16))) __bf16        v16bf;
typedef __attribute__((ext_vector_type(2)))  __bf16        v2bf;
typedef __attribute__((ext_vector_type(8)))  float         v8f;
typedef __attribute__((ext_vector_type(4)))  unsigned int  u32x4;

union Frag16 {
  v16bf          bf;
  unsigned short us[16];
  u32x4          u4[2];
};

__device__ __forceinline__ unsigned short f2bf(float f) {
  unsigned int u = __builtin_bit_cast(unsigned int, f);
  u += 0x7FFFu + ((u >> 16) & 1u);           // round-to-nearest-even
  return (unsigned short)(u >> 16);
}
// Pack two f32 -> two bf16 in one op when v_cvt_pk_bf16_f32 is exposed.
__device__ __forceinline__ unsigned int packbf(float lo, float hi) {
#if __has_builtin(__builtin_amdgcn_cvt_pk_bf16_f32)
  v2bf p = __builtin_amdgcn_cvt_pk_bf16_f32(lo, hi);
  return __builtin_bit_cast(unsigned int, p);
#else
  return (unsigned int)f2bf(lo) | ((unsigned int)f2bf(hi) << 16);
#endif
}

// Fast gate activations: v_exp_f32 + v_rcp_f32 TRANS ops only, and CDNA5's
// native v_tanh_f32 when the toolchain exposes it.
__device__ __forceinline__ float sigmoid_f(float x) {
  return __builtin_amdgcn_rcpf(1.0f + __expf(-x));
}
__device__ __forceinline__ float tanh_f(float x) {
#if __has_builtin(__builtin_amdgcn_tanhf)
  return __builtin_amdgcn_tanhf(x);          // v_tanh_f32 (TRANS)
#else
  return 2.0f * sigmoid_f(2.0f * x) - 1.0f;
#endif
}

// ---------------------------------------------------------------------------
// Kernel 1a: build bf16 B-operand for the dense stage: dwpt[col][k] (col
// padded 5->16 with zeros), contiguous in k so B-fragment loads are b128s.
// ---------------------------------------------------------------------------
__global__ __launch_bounds__(256)
void prep_dense_kernel(const float* __restrict__ dW, unsigned short* __restrict__ dwpt) {
  int idx = blockIdx.x * blockDim.x + threadIdx.x;
  if (idx >= 16 * NH) return;
  int col = idx / NH, k = idx % NH;
  float v = (col < DIN) ? dW[k * DIN + col] : 0.0f;
  dwpt[(size_t)col * NH + k] = f2bf(v);
}

// ---------------------------------------------------------------------------
// Kernel 1b: one-time X f32 -> bf16 conversion so the recurrent critical path
// does zero conversion math for x_t.
// ---------------------------------------------------------------------------
__global__ __launch_bounds__(256)
void prep_x_kernel(const float* __restrict__ X, unsigned short* __restrict__ xbf) {
  int idx = blockIdx.x * blockDim.x + threadIdx.x;
  if (idx >= XELTS) return;
  xbf[idx] = f2bf(X[idx]);
}

// ---------------------------------------------------------------------------
// Kernel 2: fused LSTM scan. One block per LSTM n; 7 waves = 7 independent
// 16-row batch tiles. Per step: one K=32 WMMA pass computes x@W + h@U for all
// 80 gate columns (K layout: 0..4 = x_t, 5..24 = h_prev, 25..31 = 0).
// Gate phase processes (j, j+1) pairs: paired LDS b64 traffic, packed bf16
// h stores (LDS b32 + global b32). No barriers inside the T loop.
// ---------------------------------------------------------------------------
__global__ __launch_bounds__(224)
void lstm_scan_kernel(const unsigned short* __restrict__ xbf, // [B,T,D] bf16
                      const float* __restrict__ Wp,   // [N,D,4H]
                      const float* __restrict__ Up,   // [N,H,4H]
                      const float* __restrict__ bp,   // [N,4H]
                      unsigned short* __restrict__ hs)// [ROWS, NH] bf16
{
  const int n    = blockIdx.x;
  const int tid  = threadIdx.x;
  const int w    = tid >> 5;      // wave / M-tile id (0..6)
  const int lane = tid & 31;
  const int nc   = lane & 15;     // column-within-tile (B/C layout) & A row id

  __shared__ float        z_lds[7][16][G4H];      // gate pre-activations
  __shared__ float        c_lds[7][16][HID];      // cell state (f32)
  __shared__ unsigned int h_lds[7][16][HID / 2];  // hidden state (2xbf16 packed)
  __shared__ float        bias_lds[G4H];

  if (tid < G4H) bias_lds[tid] = bp[n * G4H + tid];
  for (int i = lane; i < 16 * (HID / 2); i += 32) {
    h_lds[w][i / (HID / 2)][i % (HID / 2)] = 0;   // two bf16 +0.0
  }
  for (int i = lane; i < 16 * HID; i += 32) {
    c_lds[w][i / HID][i % HID] = 0.0f;
  }
  __syncthreads();                                // only barrier (bias visibility)

  // B fragments (Wcat = [W;U] padded to K=32), resident in VGPRs for all T.
  // 32x16 bf16 B layout: lane<16 holds K=0..7,16..23, lane>=16 holds K=8..15,24..31; col = lane%16.
  Frag16 bfr[5];
#pragma unroll
  for (int g = 0; g < 5; ++g) {
    int col = 16 * g + nc;
#pragma unroll
    for (int e = 0; e < 16; ++e) {
      int k = ((e < 8) ? e : e + 8) + ((lane < 16) ? 0 : 8);
      float v = 0.0f;
      if (k < DIN)            v = Wp[((size_t)n * DIN + k) * G4H + col];
      else if (k < DIN + HID) v = Up[((size_t)n * HID + (k - DIN)) * G4H + col];
      bfr[g].us[e] = f2bf(v);
    }
  }

  for (int t = 0; t < SEQT; ++t) {
    // --- assemble A fragment (16x32 bf16, documented CDNA5 layout) ---
    const unsigned short* hrow = (const unsigned short*)&h_lds[w][nc][0];
    Frag16 a;
    if (lane < 16) {                       // holds K=0..7 and K=16..23 of row `nc`
      int b = w * 16 + nc;
      const unsigned short* xp = xbf + ((size_t)((b < BATCH) ? b : 0) * SEQT + t) * DIN;
      if (t + 1 < SEQT) __builtin_prefetch(xp + DIN, 0, 0);   // next step's x
#pragma unroll
      for (int d = 0; d < DIN; ++d) a.us[d] = xp[d];          // K=0..4 : x_t (bf16)
#pragma unroll
      for (int e = 5; e < 8; ++e)  a.us[e] = hrow[e - 5];     // K=5..7  -> h0..2
#pragma unroll
      for (int e = 8; e < 16; ++e) a.us[e] = hrow[e + 3];     // K=16..23-> h11..18
    } else {                               // holds K=8..15 and K=24..31 of row `nc`
#pragma unroll
      for (int e = 0; e < 8; ++e)  a.us[e] = hrow[e + 3];     // K=8..15 -> h3..10
      a.us[8] = hrow[19];                                      // K=24    -> h19
#pragma unroll
      for (int e = 9; e < 16; ++e) a.us[e] = 0;                // K=25..31: pad
    }

    // --- z = A x Wcat : 5 column tiles of v_wmma_f32_16x16x32_bf16 ---
#pragma unroll
    for (int g = 0; g < 5; ++g) {
      v8f acc = {0.f, 0.f, 0.f, 0.f, 0.f, 0.f, 0.f, 0.f};
      acc = __builtin_amdgcn_wmma_f32_16x16x32_bf16(false, a.bf, false, bfr[g].bf,
                                                    (short)0, acc, false, false);
      int col   = 16 * g + nc;
      int rbase = (lane < 16) ? 0 : 8;
#pragma unroll
      for (int r = 0; r < 8; ++r)
        z_lds[w][rbase + r][col] = acc[r];   // C layout: lane=col, vgpr=row
    }

    // --- gates + state update: 160 (j,j+1) pairs over 32 lanes = 5 each ---
#pragma unroll
    for (int i = 0; i < 5; ++i) {
      int p   = lane + 32 * i;        // pair id 0..159
      int row = p / (HID / 2);
      int j0  = (p % (HID / 2)) * 2;  // even j -> pair never crosses a row
      float zi0 = z_lds[w][row][j0]               + bias_lds[j0];
      float zi1 = z_lds[w][row][j0 + 1]           + bias_lds[j0 + 1];
      float zf0 = z_lds[w][row][HID + j0]         + bias_lds[HID + j0];
      float zf1 = z_lds[w][row][HID + j0 + 1]     + bias_lds[HID + j0 + 1];
      float zg0 = z_lds[w][row][2 * HID + j0]     + bias_lds[2 * HID + j0];
      float zg1 = z_lds[w][row][2 * HID + j0 + 1] + bias_lds[2 * HID + j0 + 1];
      float zo0 = z_lds[w][row][3 * HID + j0]     + bias_lds[3 * HID + j0];
      float zo1 = z_lds[w][row][3 * HID + j0 + 1] + bias_lds[3 * HID + j0 + 1];
      float c0 = sigmoid_f(zf0) * c_lds[w][row][j0]     + sigmoid_f(zi0) * tanh_f(zg0);
      float c1 = sigmoid_f(zf1) * c_lds[w][row][j0 + 1] + sigmoid_f(zi1) * tanh_f(zg1);
      c_lds[w][row][j0]     = c0;
      c_lds[w][row][j0 + 1] = c1;
      float h0 = sigmoid_f(zo0) * tanh_f(c0);
      float h1 = sigmoid_f(zo1) * tanh_f(c1);
      unsigned int hp = packbf(h0, h1);
      h_lds[w][row][j0 >> 1] = hp;
      int b = w * 16 + row;
      if (b < BATCH)  // aligned 4B store (j0 even)
        *(unsigned int*)(hs + ((size_t)b * SEQT + t) * NH + n * HID + j0) = hp;
    }
  }
}

// ---------------------------------------------------------------------------
// Kernel 3: out[b,t,:] = hs[b*T+t, :] @ dense_W + dense_b
// 1600 M-tiles x 125 K-steps of v_wmma_f32_16x16x32_bf16 (N padded 5->16).
// ---------------------------------------------------------------------------
__global__ __launch_bounds__(256)
void dense_kernel(const unsigned short* __restrict__ hs,    // [ROWS, NH] bf16
                  const unsigned short* __restrict__ dwpt,  // [16, NH]   bf16
                  const float* __restrict__ db,             // [D]
                  float* __restrict__ out)                  // [ROWS, D]
{
  const int lane  = threadIdx.x & 31;
  const int wave  = threadIdx.x >> 5;
  const int mtile = blockIdx.x * 8 + wave;
  const int nc    = lane & 15;
  const size_t arowoff = (size_t)(mtile * 16 + nc) * NH;
  const int koffA0 = (lane < 16) ? 0 : 8;     // A: elements 0..7 K offset
  const int koffA1 = koffA0 + 16;             // A: elements 8..15 K offset
  const int koffB  = (lane < 16) ? 0 : 8;     // B: elements 0..7 K offset (8..15 at +16)
  const unsigned short* bcol = dwpt + (size_t)nc * NH;

  v8f acc = {0.f, 0.f, 0.f, 0.f, 0.f, 0.f, 0.f, 0.f};
  for (int ks = 0; ks < NH; ks += 32) {
    Frag16 a, b;
    a.u4[0] = *(const u32x4*)(hs + arowoff + ks + koffA0);  // global_load_b128
    a.u4[1] = *(const u32x4*)(hs + arowoff + ks + koffA1);
    b.u4[0] = *(const u32x4*)(bcol + ks + koffB);
    b.u4[1] = *(const u32x4*)(bcol + ks + koffB + 16);
    acc = __builtin_amdgcn_wmma_f32_16x16x32_bf16(false, a.bf, false, b.bf,
                                                  (short)0, acc, false, false);
  }
  if (nc < DIN) {
    float bias = db[nc];
    int rbase = mtile * 16 + ((lane < 16) ? 0 : 8);
#pragma unroll
    for (int r = 0; r < 8; ++r)
      out[(size_t)(rbase + r) * DIN + nc] = acc[r] + bias;
  }
}

// ---------------------------------------------------------------------------
// Workspace layout:
//   [0,      128KB)        : dwpt (16 x 4000 bf16, padded dense_W^T)
//   [128KB,  384KB)        : xbf  (B*T*D bf16 input)
//   [384KB,  384KB+204.8MB): hs   (25600 x 4000 bf16 hidden-state matrix)
// ---------------------------------------------------------------------------
extern "C" void kernel_launch(void* const* d_in, const int* in_sizes, int n_in,
                              void* d_out, int out_size, void* d_ws, size_t ws_size,
                              hipStream_t stream) {
  const float* X  = (const float*)d_in[0];   // inputs  [B,T,D]
  const float* W  = (const float*)d_in[1];   // W       [N,D,4H]
  const float* U  = (const float*)d_in[2];   // U       [N,H,4H]
  const float* b  = (const float*)d_in[3];   // b       [N,4H]
  const float* dW = (const float*)d_in[4];   // dense_W [N*H, D]
  const float* db = (const float*)d_in[5];   // dense_b [D]
  float* out = (float*)d_out;

  unsigned short* dwpt = (unsigned short*)d_ws;
  unsigned short* xbf  = (unsigned short*)((char*)d_ws + (size_t)131072);
  unsigned short* hs   = (unsigned short*)((char*)d_ws + (size_t)393216);

  prep_dense_kernel<<<(16 * NH + 255) / 256, 256, 0, stream>>>(dW, dwpt);
  prep_x_kernel<<<(XELTS + 255) / 256, 256, 0, stream>>>(X, xbf);
  lstm_scan_kernel<<<NLSTM, 224, 0, stream>>>(xbf, W, U, b, hs);
  dense_kernel<<<ROWS / 16 / 8, 256, 0, stream>>>(hs, dwpt, db, out);
}
